// TFMRModule_88802743812376
// MI455X (gfx1250) — matrix operand order repaired and checked by reading, not compiled, not run on hardware
//
#include <hip/hip_runtime.h>

// ---------------------------------------------------------------------------
// Problem constants (from the reference)
// ---------------------------------------------------------------------------
#define NPTS 2048
#define BB   2
#define KNN  64
#define RAD2 0.09f   // 0.3^2

typedef __attribute__((ext_vector_type(16))) _Float16 v16h;
typedef __attribute__((ext_vector_type(8)))  _Float16 v8h;
typedef __attribute__((ext_vector_type(4)))  _Float16 v4h;
typedef __attribute__((ext_vector_type(8)))  float    v8f;

union F16x16 { v16h v; v8h p[2]; };

// ---------------------------------------------------------------------------
// WMMA GEMM:  Out[b][o][m] = sum_c A(o,c) * (X1 - X2)[b][c][m] + bias[o]
//   A element (o,c) = TRANSA ? A[c*lda + o] : A[o*lda + c]
// Block: 256 threads = 8 wave32.  Tile: 64(O) x 128(M).  K-step: 32.
// Each wave owns a 16x64 strip: one A fragment, four B fragments, 4 WMMAs.
// A tile staged as Ah[o][k]; B tile staged TRANSPOSED as Bt[m][k] so each
// lane's 16 K-values are two 16-byte runs (k 8h..8h+7 and 16+8h..23+8h) -> two ds_load_b128 per frag.
// Row pitch 56 halves = 112 B (multiple of 16 B: aligned b128/b64 access,
// ~2-way bank conflicts).  M must be a multiple of 128 (true for all calls).
// GUARD handles ragged O (48) / K (6, 48) edges using clamp-address +
// select-zero so loads stay unconditional (no exec-mask serialization).
// ---------------------------------------------------------------------------
template<bool TRANSA, bool HASX2, bool GUARD>
__global__ __launch_bounds__(256)
void gemm_wmma_kernel(const float* __restrict__ A, int lda, long strideAb,
                      const float* __restrict__ X1, const float* __restrict__ X2,
                      long strideXb,
                      const float* __restrict__ bias,
                      float* __restrict__ Out, long strideOb,
                      int O, int K, int M)
{
    __shared__ __align__(16) _Float16 Ah[64][56];    // [o][k]
    __shared__ __align__(16) _Float16 Bt[128][56];   // [m][k]  (transposed)

    const int b     = blockIdx.z;
    const int oTile = blockIdx.y * 64;
    const int mTile = blockIdx.x * 128;

    const float* Ab  = A  + (long)b * strideAb;
    const float* X1b = X1 + (long)b * strideXb;
    const float* X2b = HASX2 ? (X2 + (long)b * strideXb) : nullptr;
    float*       Ob  = Out + (long)b * strideOb;

    const int tid    = threadIdx.x;
    const int wid    = tid >> 5;           // 0..7
    const int lane   = tid & 31;
    const int oSub   = (wid >> 1) * 16;    // 0,16,32,48
    const int mSub   = (wid & 1) * 64;     // 0 or 64
    const int lrow   = lane & 15;
    const int hiHalf = lane >> 4;          // 0 or 1

    v8f acc[4] = {};

    for (int kk = 0; kk < K; kk += 32) {
        // ---- stage A: 64 x 32 ----
        if (!TRANSA) {
            #pragma unroll
            for (int i = 0; i < 2; ++i) {
                int q  = i * 256 + tid;    // 512 float4 units
                int r  = q >> 3;           // 0..63
                int c4 = q & 7;
                int o  = oTile + r, k0 = kk + c4 * 4;
                v4h hv;
                if (!GUARD) {
                    const float4 f = *(const float4*)(Ab + (long)o * lda + k0);
                    hv = (v4h){(_Float16)f.x, (_Float16)f.y, (_Float16)f.z, (_Float16)f.w};
                } else {
                    // clamped addresses + select: loads stay unconditional
                    const int oc = (o < O) ? o : 0;
                    #pragma unroll
                    for (int j = 0; j < 4; ++j) {
                        int k  = k0 + j;
                        int kc = (k < K) ? k : 0;
                        float v = Ab[(long)oc * lda + kc];
                        v = (o < O && k < K) ? v : 0.0f;
                        hv[j] = (_Float16)v;
                    }
                }
                *(v4h*)&Ah[r][c4 * 4] = hv;   // ds_store_b64
            }
        } else {
            // A(o,c) = Ab[c*lda + o]; coalesce over o, scatter u16 into LDS
            #pragma unroll
            for (int i = 0; i < 8; ++i) {
                int s = i * 256 + tid;
                int r = s & 63, c = s >> 6;
                int o = oTile + r, k = kk + c;
                const int oc = (!GUARD || o < O) ? o : 0;
                const int kc = (!GUARD || k < K) ? k : 0;
                float v = Ab[(long)kc * lda + oc];
                if (GUARD) v = (o < O && k < K) ? v : 0.0f;
                Ah[r][c] = (_Float16)v;
            }
        }
        // ---- stage B: 32 x 128, stored transposed ----
        #pragma unroll
        for (int i = 0; i < 4; ++i) {
            int q  = i * 256 + tid;        // 1024 units of (m, 4 k-rows)
            int m  = q & 127;
            int r0 = (q >> 7) * 4;         // 0,4,...,28
            v4h hv;
            #pragma unroll
            for (int j = 0; j < 4; ++j) {
                int k  = kk + r0 + j;
                int kc = (!GUARD || k < K) ? k : 0;
                float v = X1b[(long)kc * M + mTile + m];
                if (HASX2) v -= X2b[(long)kc * M + mTile + m];
                if (GUARD) v = (k < K) ? v : 0.0f;
                hv[j] = (_Float16)v;
            }
            *(v4h*)&Bt[m][r0] = hv;        // ds_store_b64
        }
        // prefetch next B K-slab (steady state only)
        if (!GUARD && (kk + 32) < K)
            __builtin_prefetch(X1b + (long)(kk + 32 + (tid & 31)) * M + mTile + ((tid >> 5) << 4), 0, 1);

        __syncthreads();

        // ---- A fragment: lanes 0-15 K base 0, lanes 16-31 K base 8 ----
        F16x16 af;
        {
            const int row = oSub + lrow;
            const int kb  = hiHalf * 8;
            af.p[0] = *(const v8h*)&Ah[row][kb];        // ds_load_b128
            af.p[1] = *(const v8h*)&Ah[row][kb + 16];   // ds_load_b128
        }
        // ---- 4 B fragments + 4 WMMAs ----
        const int kbB = hiHalf * 8;
        #pragma unroll
        for (int t = 0; t < 4; ++t) {
            F16x16 bf;
            const int c0 = mSub + t * 16 + lrow;
            bf.p[0] = *(const v8h*)&Bt[c0][kbB];        // ds_load_b128
            bf.p[1] = *(const v8h*)&Bt[c0][kbB + 16];   // ds_load_b128
            acc[t] = __builtin_amdgcn_wmma_f32_16x16x32_f16(false, af.v, false, bf.v,
                                                            (short)0, acc[t], false, false);
        }
        __syncthreads();
    }

    // ---- epilogue: lane col = lane%16, rows r + 8*hiHalf ----
    #pragma unroll
    for (int r = 0; r < 8; ++r) {
        int o = oTile + oSub + r + hiHalf * 8;
        if (!GUARD || o < O) {
            float bv = bias ? bias[o] : 0.0f;
            #pragma unroll
            for (int t = 0; t < 4; ++t) {
                int m = mTile + mSub + t * 16 + lrow;
                Ob[(long)o * M + m] = acc[t][r] + bv;
            }
        }
    }
}

// ---------------------------------------------------------------------------
// Ball query: first KNN indices (ascending) with d^2 <= r^2, padded with the
// first hit.  One thread per point; candidates staged through LDS tiles.
// ---------------------------------------------------------------------------
__global__ __launch_bounds__(256)
void ball_query_kernel(const float* __restrict__ xyz, int* __restrict__ idx)
{
    __shared__ float sx[256], sy[256], sz[256];
    const int b = blockIdx.y;
    const int n = blockIdx.x * 256 + threadIdx.x;
    const float* base = xyz + (long)b * NPTS * 3;

    float cx = base[n * 3 + 0];
    float cy = base[n * 3 + 1];
    float cz = base[n * 3 + 2];

    int* op = idx + ((long)b * NPTS + n) * KNN;
    int cnt = 0;

    for (int c0 = 0; c0 < NPTS; c0 += 256) {
        int j = c0 + threadIdx.x;
        sx[threadIdx.x] = base[j * 3 + 0];
        sy[threadIdx.x] = base[j * 3 + 1];
        sz[threadIdx.x] = base[j * 3 + 2];
        __syncthreads();
        if (cnt < KNN) {
            for (int jj = 0; jj < 256; ++jj) {
                float dx = cx - sx[jj], dy = cy - sy[jj], dz = cz - sz[jj];
                float d  = dx * dx + dy * dy + dz * dz;
                if (d <= RAD2 && cnt < KNN) op[cnt++] = c0 + jj;
            }
        }
        __syncthreads();
    }
    int first = (cnt > 0) ? op[0] : n;
    for (int k = cnt; k < KNN; ++k) op[k] = first;
}

// ---------------------------------------------------------------------------
// Build x0[c][k][n] for one batch: c<3 = center xyz, c>=3 = neighbor - center
// ---------------------------------------------------------------------------
__global__ __launch_bounds__(256)
void build_x0_kernel(const float* __restrict__ xyz, const int* __restrict__ idx,
                     float* __restrict__ x0, int b)
{
    int flat = blockIdx.x * 256 + threadIdx.x;
    if (flat >= KNN * NPTS) return;
    int k = flat >> 11;
    int n = flat & (NPTS - 1);
    const float* base = xyz + (long)b * NPTS * 3;
    int j = idx[((long)b * NPTS + n) * KNN + k];
    float cx = base[n * 3 + 0], cy = base[n * 3 + 1], cz = base[n * 3 + 2];
    float gx = base[j * 3 + 0], gy = base[j * 3 + 1], gz = base[j * 3 + 2];
    const long S  = (long)KNN * NPTS;
    const long sp = (long)k * NPTS + n;
    x0[0 * S + sp] = cx;      x0[1 * S + sp] = cy;      x0[2 * S + sp] = cz;
    x0[3 * S + sp] = gx - cx; x0[4 * S + sp] = gy - cy; x0[5 * S + sp] = gz - cz;
}

// ---------------------------------------------------------------------------
// GroupNorm statistics: one block per (group, batch)
// ---------------------------------------------------------------------------
__global__ __launch_bounds__(256)
void gn_stats_kernel(const float* __restrict__ x, long strideB, int C, int G,
                     long spatial, float* __restrict__ stats)
{
    const int g = blockIdx.x, b = blockIdx.y;
    const int cg = C / G;
    const long cnt = (long)cg * spatial;
    const float* xg = x + (long)b * strideB + (long)g * cg * spatial;

    float s = 0.f, s2 = 0.f;
    for (long i = threadIdx.x; i < cnt; i += 256) {
        float v = xg[i];
        s += v; s2 += v * v;
    }
    __shared__ float ss[256], ss2[256];
    ss[threadIdx.x] = s; ss2[threadIdx.x] = s2;
    __syncthreads();
    for (int st = 128; st > 0; st >>= 1) {
        if (threadIdx.x < st) {
            ss[threadIdx.x]  += ss[threadIdx.x + st];
            ss2[threadIdx.x] += ss2[threadIdx.x + st];
        }
        __syncthreads();
    }
    if (threadIdx.x == 0) {
        float mean = ss[0] / (float)cnt;
        float var  = ss2[0] / (float)cnt - mean * mean;
        stats[(b * G + g) * 2 + 0] = mean;
        stats[(b * G + g) * 2 + 1] = var;
    }
}

// ---------------------------------------------------------------------------
// GroupNorm apply (+optional activation, residual add, mirror copy)
//   act: 0 = none, 1 = relu, 2 = leaky(0.2);  out = resid + act(gn(x))
// ---------------------------------------------------------------------------
__global__ __launch_bounds__(256)
void gn_apply_kernel(const float* __restrict__ x, long sxB,
                     const float* __restrict__ resid, long srB,
                     float* __restrict__ out, long soB,
                     float* __restrict__ copy2, long scB,
                     const float* __restrict__ stats,
                     const float* __restrict__ w, const float* __restrict__ bv,
                     int C, int G, long spatial, int act)
{
    const long total = (long)C * spatial;
    const long i = (long)blockIdx.x * 256 + threadIdx.x;
    const int b = blockIdx.y;
    if (i >= total) return;
    int c = (int)(i / spatial);
    int g = c / (C / G);
    float mean = stats[(b * G + g) * 2 + 0];
    float var  = stats[(b * G + g) * 2 + 1];
    float v = (x[(long)b * sxB + i] - mean) * rsqrtf(var + 1e-5f) * w[c] + bv[c];
    if (act == 1)      v = fmaxf(v, 0.0f);
    else if (act == 2) v = (v > 0.0f) ? v : 0.2f * v;
    if (resid) v += resid[(long)b * srB + i];
    out[(long)b * soB + i] = v;
    if (copy2) copy2[(long)b * scB + i] = v;
}

// ---------------------------------------------------------------------------
// Max over K axis: x[192][64][2048] -> f[192][2048]  (one batch)
// ---------------------------------------------------------------------------
__global__ __launch_bounds__(256)
void maxk_kernel(const float* __restrict__ x, float* __restrict__ f)
{
    int i = blockIdx.x * 256 + threadIdx.x;
    if (i >= 192 * NPTS) return;
    int c = i >> 11, n = i & (NPTS - 1);
    const float* p = x + (long)c * KNN * NPTS + n;
    float m = -3.4e38f;
    for (int k = 0; k < KNN; ++k) m = fmaxf(m, p[(long)k * NPTS]);
    f[i] = m;
}

// ---------------------------------------------------------------------------
// Row softmax over att[B*N][N] (in place)
// ---------------------------------------------------------------------------
__global__ __launch_bounds__(256)
void softmax_row_kernel(float* __restrict__ att)
{
    float* p = att + (long)blockIdx.x * NPTS;
    __shared__ float red[256];

    float mx = -3.4e38f;
    for (int i = threadIdx.x; i < NPTS; i += 256) mx = fmaxf(mx, p[i]);
    red[threadIdx.x] = mx;
    __syncthreads();
    for (int st = 128; st > 0; st >>= 1) {
        if (threadIdx.x < st) red[threadIdx.x] = fmaxf(red[threadIdx.x], red[threadIdx.x + st]);
        __syncthreads();
    }
    mx = red[0];
    __syncthreads();

    float s = 0.f;
    for (int i = threadIdx.x; i < NPTS; i += 256) {
        float e = expf(p[i] - mx);
        p[i] = e; s += e;
    }
    red[threadIdx.x] = s;
    __syncthreads();
    for (int st = 128; st > 0; st >>= 1) {
        if (threadIdx.x < st) red[threadIdx.x] += red[threadIdx.x + st];
        __syncthreads();
    }
    float inv = 1.0f / red[0];
    for (int i = threadIdx.x; i < NPTS; i += 256) p[i] *= inv;
}

// Column sums over rows (axis=1 of att[b][n][m]) and rescale
__global__ __launch_bounds__(256)
void colsum_kernel(const float* __restrict__ att, float* __restrict__ cs)
{
    int b = blockIdx.y;
    int m = blockIdx.x * 256 + threadIdx.x;
    const float* p = att + (long)b * NPTS * NPTS + m;
    float s = 0.f;
    for (int n = 0; n < NPTS; ++n) s += p[(long)n * NPTS];
    cs[b * NPTS + m] = s;
}

__global__ __launch_bounds__(256)
void colscale_kernel(float* __restrict__ att, const float* __restrict__ cs)
{
    int b = blockIdx.y;
    long i = (long)blockIdx.x * 256 + threadIdx.x;
    if (i >= (long)NPTS * NPTS) return;
    int m = (int)(i & (NPTS - 1));
    att[(long)b * NPTS * NPTS + i] /= (1e-9f + cs[b * NPTS + m]);
}

// ---------------------------------------------------------------------------
// Final: transpose to (B, N, 768) and L2-normalize each point
// ---------------------------------------------------------------------------
__global__ __launch_bounds__(256)
void finalize_kernel(const float* __restrict__ y, float* __restrict__ out)
{
    const int b = blockIdx.y, n = blockIdx.x;
    const float* p = y + (long)b * 768 * NPTS + n;
    __shared__ float red[256];
    float s = 0.f;
    for (int c = threadIdx.x; c < 768; c += 256) {
        float v = p[(long)c * NPTS];
        s += v * v;
    }
    red[threadIdx.x] = s;
    __syncthreads();
    for (int st = 128; st > 0; st >>= 1) {
        if (threadIdx.x < st) red[threadIdx.x] += red[threadIdx.x + st];
        __syncthreads();
    }
    float inv = 1.0f / (sqrtf(red[0]) + 1e-8f);
    float* o = out + ((long)b * NPTS + n) * 768;
    for (int c = threadIdx.x; c < 768; c += 256) o[c] = p[(long)c * NPTS] * inv;
}

// ---------------------------------------------------------------------------
// Host orchestration
// ---------------------------------------------------------------------------
extern "C" void kernel_launch(void* const* d_in, const int* in_sizes, int n_in,
                              void* d_out, int out_size, void* d_ws, size_t ws_size,
                              hipStream_t stream)
{
    (void)in_sizes; (void)n_in; (void)out_size; (void)ws_size;

    const float* src = (const float*)d_in[0];
    const float* tgt = (const float*)d_in[1];
    const float* lf_w[3]  = {(const float*)d_in[5],  (const float*)d_in[8],  (const float*)d_in[11]};
    const float* lf_gw[3] = {(const float*)d_in[6],  (const float*)d_in[9],  (const float*)d_in[12]};
    const float* lf_gb[3] = {(const float*)d_in[7],  (const float*)d_in[10], (const float*)d_in[13]};
    const float* qk_w   = (const float*)d_in[14];
    const float* v_w    = (const float*)d_in[15];
    const float* v_b    = (const float*)d_in[16];
    const float* t_w    = (const float*)d_in[17];
    const float* t_b    = (const float*)d_in[18];
    const float* blk_gw = (const float*)d_in[19];
    const float* blk_gb = (const float*)d_in[20];
    const float* fuse_w  = (const float*)d_in[21];
    const float* fuse_gw = (const float*)d_in[22];
    const float* fuse_gb = (const float*)d_in[23];

    // ---- workspace carve-up ----
    char* ws = (char*)d_ws;
    size_t off = 0;
    int* idxbuf = (int*)(ws + off);          off += (size_t)BB * NPTS * KNN * 4;
    off = (off + 255) & ~(size_t)255;
    float* R0 = (float*)(ws + off);          off += (size_t)256 * KNN * NPTS * 4;  // 134 MB arena
    float* R1 = (float*)(ws + off);          off += (size_t)192 * KNN * NPTS * 4;  // 100 MB arena
    float* xf    = (float*)(ws + off);       off += (size_t)BB * 192 * NPTS * 4;
    float* qkb   = (float*)(ws + off);       off += (size_t)BB * 48  * NPTS * 4;
    float* xv    = (float*)(ws + off);       off += (size_t)BB * 192 * NPTS * 4;
    float* xr    = (float*)(ws + off);       off += (size_t)BB * 192 * NPTS * 4;
    float* tbf   = (float*)(ws + off);       off += (size_t)BB * 192 * NPTS * 4;
    float* cs    = (float*)(ws + off);       off += (size_t)BB * NPTS * 4;
    float* feats = (float*)(ws + off);       off += (size_t)BB * 768 * NPTS * 4;
    float* fuseo = (float*)(ws + off);       off += (size_t)BB * 768 * NPTS * 4;
    float* stats = (float*)(ws + off);       off += 256 * 4;

    const long SPC = (long)KNN * NPTS;       // conv spatial = 131072

    auto gemm = [&](int transA, int hasX2,
                    const float* A, int lda, long sAb,
                    const float* X1, const float* X2, long sXb,
                    const float* bias, float* Out, long sOb,
                    int O, int K, int M, int batch) {
        dim3 g(M / 128, (O + 63) / 64, batch);
        const bool guard = (O % 64) || (K % 32);
        if (transA) {
            hipLaunchKernelGGL((gemm_wmma_kernel<true, false, true>), g, dim3(256), 0, stream,
                               A, lda, sAb, X1, X2, sXb, bias, Out, sOb, O, K, M);
        } else if (hasX2) {
            hipLaunchKernelGGL((gemm_wmma_kernel<false, true, false>), g, dim3(256), 0, stream,
                               A, lda, sAb, X1, X2, sXb, bias, Out, sOb, O, K, M);
        } else if (guard) {
            hipLaunchKernelGGL((gemm_wmma_kernel<false, false, true>), g, dim3(256), 0, stream,
                               A, lda, sAb, X1, X2, sXb, bias, Out, sOb, O, K, M);
        } else {
            hipLaunchKernelGGL((gemm_wmma_kernel<false, false, false>), g, dim3(256), 0, stream,
                               A, lda, sAb, X1, X2, sXb, bias, Out, sOb, O, K, M);
        }
    };
    auto gnorm1 = [&](float* x, int C, int G, long spatial,
                      const float* w, const float* b) {
        hipLaunchKernelGGL(gn_stats_kernel, dim3(G, 1), dim3(256), 0, stream,
                           x, 0L, C, G, spatial, stats);
        long tot = (long)C * spatial;
        hipLaunchKernelGGL(gn_apply_kernel, dim3((unsigned)((tot + 255) / 256), 1), dim3(256), 0, stream,
                           x, 0L, (const float*)nullptr, 0L, x, 0L,
                           (float*)nullptr, 0L, stats, w, b, C, G, spatial, 1);
    };

    for (int br = 0; br < 2; ++br) {
        const float* xyz = br ? tgt : src;

        // ---- local feature stage (per batch to bound workspace) ----
        hipLaunchKernelGGL(ball_query_kernel, dim3(NPTS / 256, BB), dim3(256), 0, stream,
                           xyz, idxbuf);
        for (int b = 0; b < BB; ++b) {
            hipLaunchKernelGGL(build_x0_kernel, dim3((KNN * NPTS) / 256), dim3(256), 0, stream,
                               xyz, idxbuf, R0, b);
            // L1: 6 -> 128  (x0 in R0 -> act1 in R1)
            gemm(0, 0, lf_w[0], 6, 0, R0, nullptr, 0, nullptr, R1, 0, 128, 6, (int)SPC, 1);
            gnorm1(R1, 128, 32, SPC, lf_gw[0], lf_gb[0]);
            // L2: 128 -> 256 (R1 -> R0)
            gemm(0, 0, lf_w[1], 128, 0, R1, nullptr, 0, nullptr, R0, 0, 256, 128, (int)SPC, 1);
            gnorm1(R0, 256, 32, SPC, lf_gw[1], lf_gb[1]);
            // L3: 256 -> 192 (R0 -> R1)
            gemm(0, 0, lf_w[2], 256, 0, R0, nullptr, 0, nullptr, R1, 0, 192, 256, (int)SPC, 1);
            gnorm1(R1, 192, 32, SPC, lf_gw[2], lf_gb[2]);
            // max over K -> xf[b]
            hipLaunchKernelGGL(maxk_kernel, dim3((192 * NPTS + 255) / 256), dim3(256), 0, stream,
                               R1, xf + (long)b * 192 * NPTS);
        }

        // ---- attention stage (full batch; att matrix reuses R0) ----
        const long sX   = (long)192 * NPTS;
        const long sQK  = (long)48 * NPTS;
        const long sAtt = (long)NPTS * NPTS;
        for (int i = 0; i < 4; ++i) {
            // qk = qkw @ x  (48 x 2048 per batch)
            gemm(0, 0, qk_w + (long)i * 48 * 192, 192, 0,
                 xf, nullptr, sX, nullptr, qkb, sQK, 48, 192, NPTS, BB);
            // att = qk^T @ qk  (Gram, 2048 x 2048)
            gemm(1, 0, qkb, NPTS, sQK,
                 qkb, nullptr, sQK, nullptr, R0, sAtt, NPTS, 48, NPTS, BB);
            hipLaunchKernelGGL(softmax_row_kernel, dim3(BB * NPTS), dim3(256), 0, stream, R0);
            hipLaunchKernelGGL(colsum_kernel, dim3(NPTS / 256, BB), dim3(256), 0, stream, R0, cs);
            hipLaunchKernelGGL(colscale_kernel,
                               dim3((unsigned)((sAtt + 255) / 256), BB), dim3(256), 0, stream, R0, cs);
            // x_v = vw @ x + vb
            gemm(0, 0, v_w + (long)i * 192 * 192, 192, 0,
                 xf, nullptr, sX, v_b + i * 192, xv, sX, 192, 192, NPTS, BB);
            // x_r = x_v @ att
            gemm(0, 0, xv, NPTS, sX,
                 R0, nullptr, sAtt, nullptr, xr, sX, 192, NPTS, NPTS, BB);
            // t = tw @ (x - x_r) + tb
            gemm(0, 1, t_w + (long)i * 192 * 192, 192, 0,
                 xf, xr, sX, t_b + i * 192, tbf, sX, 192, 192, NPTS, BB);
            // x = x + relu(gn(t));  mirror into feats slice i
            hipLaunchKernelGGL(gn_stats_kernel, dim3(32, BB), dim3(256), 0, stream,
                               tbf, sX, 192, 32, (long)NPTS, stats);
            hipLaunchKernelGGL(gn_apply_kernel,
                               dim3((unsigned)((192 * NPTS + 255) / 256), BB), dim3(256), 0, stream,
                               tbf, sX, xf, sX, xf, sX,
                               feats + (long)i * 192 * NPTS, (long)768 * NPTS,
                               stats, blk_gw + i * 192, blk_gb + i * 192,
                               192, 32, (long)NPTS, 1);
        }

        // ---- fuse + GN(16) + leaky relu ----
        gemm(0, 0, fuse_w, 768, 0,
             feats, nullptr, (long)768 * NPTS, nullptr, fuseo, (long)768 * NPTS,
             768, 768, NPTS, BB);
        hipLaunchKernelGGL(gn_stats_kernel, dim3(16, BB), dim3(256), 0, stream,
                           fuseo, (long)768 * NPTS, 768, 16, (long)NPTS, stats);
        hipLaunchKernelGGL(gn_apply_kernel,
                           dim3((unsigned)((768 * NPTS + 255) / 256), BB), dim3(256), 0, stream,
                           fuseo, (long)768 * NPTS, (const float*)nullptr, 0L,
                           fuseo, (long)768 * NPTS, (float*)nullptr, 0L,
                           stats, fuse_gw, fuse_gb, 768, 16, (long)NPTS, 2);

        // ---- transpose + L2 normalize -> d_out[branch] ----
        hipLaunchKernelGGL(finalize_kernel, dim3(NPTS, BB), dim3(256), 0, stream,
                           fuseo, (float*)d_out + (long)br * BB * NPTS * 768);
    }
}
